// FastPointTransformerLayer_64665027608654
// MI455X (gfx1250) — matrix operand order repaired and checked by reading, not compiled, not run on hardware
//
#include <hip/hip_runtime.h>
#include <hip/hip_bf16.h>
#include <math.h>

#define NPTS  100000
#define NCH   128
#define NH    8
#define HC    16
#define KV    27
#define NMAP  1600000
#define EPS   1e-5f
#define NT16  (NPTS/16)   // 6250, exact

typedef __attribute__((ext_vector_type(16))) __bf16 v16bf;
typedef __attribute__((ext_vector_type(8)))  float  v8f;

// ---- workspace byte offsets ----
#define OFF_BN1    0u                                // 8 floats (sum[3], ssum[3])
#define OFF_BN2S   64u                               // 128 floats
#define OFF_BN2Q   576u                              // 128 floats
#define OFF_BN2SC  1088u                             // 128 floats: scale
#define OFF_BN2SH  1600u                             // 128 floats: shift
#define OFF_W3P    4096u                             // 16384 bf16, fragment-packed
#define OFF_WQP    36864u
#define OFF_WVP    69632u
#define OFF_WOP    102400u
#define OFF_NPE    135168u                           // 27*128 f32
#define OFF_Y2     151552u                           // N*128 f32
#define OFF_NQ     (OFF_Y2 + (size_t)NPTS*NCH*4u)
#define OFF_V      (OFF_NQ + (size_t)NPTS*NCH*4u)

__device__ __forceinline__ v8f wmma_bf16(v16bf a, v16bf b, v8f c) {
  return __builtin_amdgcn_wmma_f32_16x16x32_bf16(false, a, false, b, (short)0, c,
                                                 false, false);
}

// Packed B fragment: lane's 16 bf16 contiguous -> one 32B vector load (2x b128).
__device__ __forceinline__ v16bf load_bfrag(const __bf16* __restrict__ Wp,
                                            int ks, int lane, int n0) {
  return *(const v16bf*)(Wp + ((((n0 << 2) + ks) * 32 + lane) << 4));
}

// Load all 4 K-step fragments first (one clause), then chain WMMAs so the
// loadcnt waits stagger instead of stalling before every WMMA.
__device__ __forceinline__ v8f gemm_tile(const v16bf* __restrict__ a,
                                         const __bf16* __restrict__ Wp,
                                         int lane, int n0) {
  v16bf bf[4];
#pragma unroll
  for (int ks = 0; ks < 4; ++ks) bf[ks] = load_bfrag(Wp, ks, lane, n0);
  v8f d = {};
#pragma unroll
  for (int ks = 0; ks < 4; ++ks) d = wmma_bf16(a[ks], bf[ks], d);
  return d;
}

// ---------------- setup kernels ----------------
__global__ void k_init(float* __restrict__ p) {
  p[threadIdx.x] = 0.0f;   // zero first 512 floats (BN accumulators)
}

// fp32 [128x128] -> bf16 in B-fragment order:
// packed[(((n0*4+ks)*32+lane)*16)+j] = W[(ks*32+(j>>3)*16+(lane>>4)*8+(j&7))*128 + n0*16+(lane&15)]
__global__ void k_wpack(const float* __restrict__ a, const float* __restrict__ b,
                        const float* __restrict__ c, const float* __restrict__ d,
                        __bf16* __restrict__ oa, __bf16* __restrict__ ob,
                        __bf16* __restrict__ oc, __bf16* __restrict__ od) {
  int i = blockIdx.x * 256 + threadIdx.x;      // 0..65535
  int m = i >> 14, e = i & 16383;
  int j = e & 15, lane = (e >> 4) & 31, ks = (e >> 9) & 3, n0 = (e >> 11) & 7;
  int row = ks * 32 + ((j >> 3) << 4) + ((lane >> 4) << 3) + (j & 7);
  int col = n0 * 16 + (lane & 15);
  const float* s = (m == 0) ? a : (m == 1) ? b : (m == 2) ? c : d;
  __bf16* t = (m == 0) ? oa : (m == 1) ? ob : (m == 2) ? oc : od;
  t[e] = (__bf16)s[row * NCH + col];
}

__global__ void k_npe(const float* __restrict__ pe, float* __restrict__ npe) {
  int t = threadIdx.x;
  if (t >= KV * NH) return;
  float v[16], s = 0.0f;
#pragma unroll
  for (int c = 0; c < 16; ++c) { v[c] = pe[t * 16 + c]; s += v[c] * v[c]; }
  float inv = 1.0f / fmaxf(sqrtf(s), 1e-12f);
#pragma unroll
  for (int c = 0; c < 16; ++c) npe[t * 16 + c] = v[c] * inv;
}

// ---------------- BatchNorm stats / tiny MLP ----------------
__global__ __launch_bounds__(256) void k_bn1(const float* __restrict__ points,
                                             const float* __restrict__ W1,
                                             float* __restrict__ bn1) {
  int n = blockIdx.x * 256 + threadIdx.x;
  float y[3] = {0.f, 0.f, 0.f};
  if (n < NPTS) {
    float p0 = points[n * 3], p1 = points[n * 3 + 1], p2 = points[n * 3 + 2];
#pragma unroll
    for (int j = 0; j < 3; ++j) y[j] = p0 * W1[j] + p1 * W1[3 + j] + p2 * W1[6 + j];
  }
#pragma unroll
  for (int j = 0; j < 3; ++j) {
    float s = y[j], q = y[j] * y[j];
    for (int off = 16; off; off >>= 1) { s += __shfl_down(s, off); q += __shfl_down(q, off); }
    if ((threadIdx.x & 31) == 0) { atomicAdd(bn1 + j, s); atomicAdd(bn1 + 3 + j, q); }
  }
}

__global__ __launch_bounds__(256) void k_bn2(
    const float* __restrict__ points, const float* __restrict__ W1,
    const float* __restrict__ g1, const float* __restrict__ b1,
    const float* __restrict__ W2, const float* __restrict__ bn1,
    float* __restrict__ y2ws, float* __restrict__ bn2s, float* __restrict__ bn2q) {
  __shared__ float ls[NCH], lq[NCH];
  int t = threadIdx.x;
  if (t < NCH) { ls[t] = 0.f; lq[t] = 0.f; }
  __syncthreads();
  int n = blockIdx.x * 256 + t;
  float h[3] = {0.f, 0.f, 0.f};
  const float invN = 1.0f / (float)NPTS;
  if (n < NPTS) {
    float p0 = points[n * 3], p1 = points[n * 3 + 1], p2 = points[n * 3 + 2];
#pragma unroll
    for (int j = 0; j < 3; ++j) {
      float y = p0 * W1[j] + p1 * W1[3 + j] + p2 * W1[6 + j];
      float mu = bn1[j] * invN;
      float var = bn1[3 + j] * invN - mu * mu;
      h[j] = fmaxf((y - mu) * rsqrtf(var + EPS) * g1[j] + b1[j], 0.0f);
    }
  }
  for (int c = 0; c < NCH; ++c) {
    float y = h[0] * W2[c] + h[1] * W2[NCH + c] + h[2] * W2[2 * NCH + c];
    if (n < NPTS) y2ws[(size_t)n * NCH + c] = y;
    float s = y, q = y * y;
    for (int off = 16; off; off >>= 1) { s += __shfl_down(s, off); q += __shfl_down(q, off); }
    if ((t & 31) == 0) { atomicAdd(&ls[c], s); atomicAdd(&lq[c], q); }  // ds_add_f32
  }
  __syncthreads();
  if (t < NCH) { atomicAdd(bn2s + t, ls[t]); atomicAdd(bn2q + t, lq[t]); }
}

// fold BN2 into per-channel scale/shift: h = relu(y*sc + sh)
__global__ void k_bnfin(const float* __restrict__ bn2s, const float* __restrict__ bn2q,
                        const float* __restrict__ g2, const float* __restrict__ b2,
                        float* __restrict__ sc, float* __restrict__ sh) {
  int c = threadIdx.x;
  if (c >= NCH) return;
  const float invN = 1.0f / (float)NPTS;
  float mu = bn2s[c] * invN;
  float var = bn2q[c] * invN - mu * mu;
  float s = rsqrtf(var + EPS) * g2[c];
  sc[c] = s;
  sh[c] = b2[c] - mu * s;
}

// ---------------- fused BN2 -> W3 -> (+feats) -> Wq/Wv WMMA ----------------
__global__ __launch_bounds__(128) void k_main(
    const float* __restrict__ feats, const float* __restrict__ y2ws,
    const float* __restrict__ sc, const float* __restrict__ sh,
    const float* __restrict__ b3,
    const __bf16* __restrict__ W3p, const __bf16* __restrict__ Wqp,
    const __bf16* __restrict__ Wvp,
    const float* __restrict__ bq, const float* __restrict__ bv,
    float* __restrict__ nqout, float* __restrict__ vout, float* __restrict__ acc) {
  __shared__ float xs[4 * 16 * NCH];        // 32 KB, one 16x128 tile per wave
  int wave = threadIdx.x >> 5;
  int lane = threadIdx.x & 31;
  int tile = blockIdx.x * 4 + wave;
  if (tile >= NT16) return;                 // wave-uniform; no block barriers used
  __builtin_prefetch(W3p, 0, 0);            // global_prefetch_b8
  __builtin_prefetch(Wqp, 0, 0);
  __builtin_prefetch(Wvp, 0, 0);
  float* xw = xs + wave * (16 * NCH);
  int r = lane & 15, half = lane >> 4;
  int koff = half * 8, rbase = half * 8;
  int grow = tile * 16 + r;

  // A fragments: h2 = relu(y2*sc + sh)   (row-major per lane, vectorizable)
  v16bf ah[4];
#pragma unroll
  for (int ks = 0; ks < 4; ++ks)
#pragma unroll
    for (int g = 0; g < 2; ++g)
#pragma unroll
      for (int j = 0; j < 8; ++j) {
        int c = ks * 32 + g * 16 + koff + j;
        float y = y2ws[(size_t)grow * NCH + c];
        ah[ks][g * 8 + j] = (__bf16)fmaxf(fmaf(y, sc[c], sh[c]), 0.0f);
      }

  // intra = h2 @ W3 + b3 -> LDS (feats is added row-major in the rebuild)
#pragma unroll
  for (int n0 = 0; n0 < 8; ++n0) {
    v8f d = gemm_tile(ah, W3p, lane, n0);
    int col = n0 * 16 + (lane & 15);
    float b3v = b3[col];
#pragma unroll
    for (int j = 0; j < 8; ++j)
      xw[(rbase + j) * NCH + col] = d[j] + b3v;
  }
  asm volatile("s_wait_dscnt 0x0" ::: "memory");   // wave-local LDS fence (lockstep wave32)

  // A fragments: x = intra + b3 + feats (bf16); feats row-major per lane
  v16bf ax[4];
#pragma unroll
  for (int ks = 0; ks < 4; ++ks)
#pragma unroll
    for (int g = 0; g < 2; ++g)
#pragma unroll
      for (int j = 0; j < 8; ++j) {
        int c = ks * 32 + g * 16 + koff + j;
        float x = xw[r * NCH + c] + feats[(size_t)grow * NCH + c];
        ax[ks][g * 8 + j] = (__bf16)x;
      }

  // q = x @ Wq + bq, l2norm per head (head == 16-col WMMA tile == 16 lanes)
#pragma unroll
  for (int n0 = 0; n0 < 8; ++n0) {
    v8f d = gemm_tile(ax, Wqp, lane, n0);
    int col = n0 * 16 + (lane & 15);
    float bqv = bq[col];
#pragma unroll
    for (int j = 0; j < 8; ++j) {
      float q = d[j] + bqv;
      float s = q * q;
      s += __shfl_xor(s, 1); s += __shfl_xor(s, 2);
      s += __shfl_xor(s, 4); s += __shfl_xor(s, 8);
      float inv = 1.0f / fmaxf(sqrtf(s), 1e-12f);
      nqout[((size_t)tile * 16 + rbase + j) * NCH + col] = q * inv;
    }
  }

  // v = x @ Wv + bv ; zero the scatter accumulator
#pragma unroll
  for (int n0 = 0; n0 < 8; ++n0) {
    v8f d = gemm_tile(ax, Wvp, lane, n0);
    int col = n0 * 16 + (lane & 15);
    float bvv = bv[col];
#pragma unroll
    for (int j = 0; j < 8; ++j) {
      size_t o = ((size_t)tile * 16 + rbase + j) * NCH + col;
      vout[o] = d[j] + bvv;
      acc[o] = 0.0f;
    }
  }
}

// ---------------- gather / attention / scatter (L2-resident) ----------------
__global__ __launch_bounds__(256) void k_scatter(
    const int* __restrict__ kq, const float* __restrict__ nq,
    const float* __restrict__ npe, const float* __restrict__ v,
    float* __restrict__ acc) {
  int t = threadIdx.x;
  long m = (long)blockIdx.x * 2 + (t >> 7);
  int ch = t & 127;                 // ch = h*16 + c ; a head = 16 contiguous lanes
  if (m >= NMAP) return;
  int kk = kq[m];
  int qi = kq[NMAP + m];
  int kern = kk % KV;
  int key = kk / KV;
  float p = nq[(size_t)qi * NCH + ch] * npe[kern * NCH + ch];
  p += __shfl_xor(p, 1); p += __shfl_xor(p, 2);
  p += __shfl_xor(p, 4); p += __shfl_xor(p, 8);   // attn[m,h], replicated in head
  float val = p * v[(size_t)key * NCH + ch];
  atomicAdd(&acc[(size_t)qi * NCH + ch], val);    // global_atomic_add_f32
}

// ---------------- out = acc @ Wo + bo (in place, row-tile safe) ----------------
__global__ __launch_bounds__(128) void k_out(float* __restrict__ acc,
                                             const __bf16* __restrict__ Wop,
                                             const float* __restrict__ bo) {
  int wave = threadIdx.x >> 5;
  int lane = threadIdx.x & 31;
  int tile = blockIdx.x * 4 + wave;
  if (tile >= NT16) return;
  int r = lane & 15, half = lane >> 4;
  int koff = half * 8, rbase = half * 8;
  v16bf a[4];
#pragma unroll
  for (int ks = 0; ks < 4; ++ks)
#pragma unroll
    for (int g = 0; g < 2; ++g)
#pragma unroll
      for (int j = 0; j < 8; ++j) {
        int c = ks * 32 + g * 16 + koff + j;
        a[ks][g * 8 + j] = (__bf16)acc[((size_t)tile * 16 + r) * NCH + c];
      }
#pragma unroll
  for (int n0 = 0; n0 < 8; ++n0) {
    v8f d = gemm_tile(a, Wop, lane, n0);
    int col = n0 * 16 + (lane & 15);
    float bov = bo[col];
#pragma unroll
    for (int j = 0; j < 8; ++j)
      acc[((size_t)tile * 16 + rbase + j) * NCH + col] = d[j] + bov;
  }
}

extern "C" void kernel_launch(void* const* d_in, const int* in_sizes, int n_in,
                              void* d_out, int out_size, void* d_ws, size_t ws_size,
                              hipStream_t stream) {
  const float* feats  = (const float*)d_in[0];
  const float* points = (const float*)d_in[1];
  const int*   kq     = (const int*)d_in[2];
  const float* W1 = (const float*)d_in[3];
  const float* g1 = (const float*)d_in[4];
  const float* b1 = (const float*)d_in[5];
  const float* W2 = (const float*)d_in[6];
  const float* g2 = (const float*)d_in[7];
  const float* b2 = (const float*)d_in[8];
  const float* W3 = (const float*)d_in[9];
  const float* b3 = (const float*)d_in[10];
  const float* Wq = (const float*)d_in[11];
  const float* bq = (const float*)d_in[12];
  const float* Wv = (const float*)d_in[13];
  const float* bv = (const float*)d_in[14];
  const float* pos = (const float*)d_in[15];
  const float* Wo = (const float*)d_in[16];
  const float* bo = (const float*)d_in[17];

  char* ws = (char*)d_ws;
  float*  bn1   = (float*)(ws + OFF_BN1);
  float*  bn2s  = (float*)(ws + OFF_BN2S);
  float*  bn2q  = (float*)(ws + OFF_BN2Q);
  float*  bn2sc = (float*)(ws + OFF_BN2SC);
  float*  bn2sh = (float*)(ws + OFF_BN2SH);
  __bf16* W3p   = (__bf16*)(ws + OFF_W3P);
  __bf16* Wqp   = (__bf16*)(ws + OFF_WQP);
  __bf16* Wvp   = (__bf16*)(ws + OFF_WVP);
  __bf16* Wop   = (__bf16*)(ws + OFF_WOP);
  float*  npe   = (float*)(ws + OFF_NPE);
  float*  y2    = (float*)(ws + OFF_Y2);
  float*  nqw   = (float*)(ws + OFF_NQ);
  float*  vw    = (float*)(ws + OFF_V);
  float*  out   = (float*)d_out;

  k_init<<<1, 512, 0, stream>>>(bn1);
  k_wpack<<<256, 256, 0, stream>>>(W3, Wq, Wv, Wo, W3p, Wqp, Wvp, Wop);
  k_npe<<<1, 256, 0, stream>>>(pos, npe);
  k_bn1<<<(NPTS + 255) / 256, 256, 0, stream>>>(points, W1, bn1);
  k_bn2<<<(NPTS + 255) / 256, 256, 0, stream>>>(points, W1, g1, b1, W2, bn1, y2, bn2s, bn2q);
  k_bnfin<<<1, 128, 0, stream>>>(bn2s, bn2q, g2, b2, bn2sc, bn2sh);
  k_main<<<(NT16 + 3) / 4, 128, 0, stream>>>(feats, y2, bn2sc, bn2sh, b3,
                                             W3p, Wqp, Wvp, bq, bv, nqw, vw, out);
  k_scatter<<<NMAP / 2, 256, 0, stream>>>(kq, nqw, npe, vw, out);
  k_out<<<(NT16 + 3) / 4, 128, 0, stream>>>(out, Wop, bo);
}
